// MultiHeadAttentionLayer_87814901334530
// MI455X (gfx1250) — matrix operand order repaired and checked by reading, not compiled, not run on hardware
//
#include <hip/hip_runtime.h>

typedef __attribute__((ext_vector_type(16))) __bf16 v16bf;
typedef __attribute__((ext_vector_type(8)))  float  v8f;

union BfPair { unsigned u; __bf16 b[2]; };

__device__ __forceinline__ unsigned short f2bfu(float f) {
    union { float f; unsigned u; } a; a.f = f;
    unsigned r = (a.u + 0x7FFFu + ((a.u >> 16) & 1u)) >> 16;
    return (unsigned short)r;
}

__device__ __forceinline__ v8f zero8() {
    v8f z;
#pragma unroll
    for (int i = 0; i < 8; ++i) z[i] = 0.0f;
    return z;
}

// A-matrix fragment (16x32 bf16), ISA layout:
// lanes 0-15 (laneHi=0): K = 0..7 (v0-3), 16..23 (v4-7); lanes 16-31: K = 8..15, 24..31.
// rowPtr: contiguous 32 bf16 of this lane's row (4B aligned, even stride).
__device__ __forceinline__ v16bf load_a_frag(const unsigned short* rowPtr, int laneHi) {
    const unsigned* r32 = (const unsigned*)rowPtr;
    int koff = laneHi * 4;  // in uint units (8 shorts)
    v16bf a;
#pragma unroll
    for (int v = 0; v < 4; ++v) {
        BfPair p0; p0.u = r32[koff + v];
        BfPair p1; p1.u = r32[8 + koff + v];
        a[2 * v]     = p0.b[0]; a[2 * v + 1]     = p0.b[1];
        a[8 + 2 * v] = p1.b[0]; a[8 + 2 * v + 1] = p1.b[1];
    }
    return a;
}

// B-matrix fragment (32x16 bf16): lane holds column n=lane&15;
// lanes 0-15 hold K=0..7,16..23; lanes 16-31 hold K=8..15,24..31 (2 per VGPR).
// colPtr: contiguous k-run for this lane's column (transposed LDS storage).
__device__ __forceinline__ v16bf load_b_frag(const unsigned short* colPtr, int laneHi) {
    const unsigned* r32 = (const unsigned*)colPtr;
    int kb = laneHi * 4;  // in uint units (8 shorts)
    v16bf b;
#pragma unroll
    for (int v = 0; v < 8; ++v) {
        BfPair p; p.u = r32[(v < 4) ? (kb + v) : (4 + kb + v)];
        b[2 * v] = p.b[0]; b[2 * v + 1] = p.b[1];
    }
    return b;
}

__device__ __forceinline__ v8f wmma_bf16(v16bf a, v16bf b, v8f c) {
    return __builtin_amdgcn_wmma_f32_16x16x32_bf16(false, a, false, b, (short)0, c, false, false);
}

// ---------------------------------------------------------------------------
// Kernel 1: fused QKV projection. Block tile 64(M) x 64(N), 4 waves.
// n0 < 256 -> Q = x@Wq+bq ; n0 in [256,768) -> KV = z@Wkv+bkv.
// Epilogue scatters bf16 results into shifted+window-partitioned layout
// [B, head, win(256), t(64), dk(32)].
// ---------------------------------------------------------------------------
__global__ __launch_bounds__(128) void swin_qkv_gemm(
    const float* __restrict__ x, const float* __restrict__ z,
    const float* __restrict__ Wq, const float* __restrict__ bq,
    const float* __restrict__ Wkv, const float* __restrict__ bkv,
    unsigned short* __restrict__ qw, unsigned short* __restrict__ kw,
    unsigned short* __restrict__ vw)
{
    const int nt = blockIdx.x % 12;
    const int mt = blockIdx.x / 12;
    const int n0 = nt * 64;
    const int m0 = mt * 64;
    const int tid = threadIdx.x;
    const int wid = tid >> 5, lane = tid & 31;
    const int laneHi = lane >> 4, laneLo = lane & 15;

    const bool isQ = (n0 < 256);
    const float* X  = isQ ? x : z;
    const float* Wp = isQ ? Wq : Wkv;
    const int   ldW = isQ ? 256 : 512;
    const int nBase = isQ ? n0 : (n0 - 256);

    __shared__ __align__(16) unsigned short Asm[64][40];  // [m][k] bf16
    __shared__ __align__(16) unsigned short Bsm[64][40];  // [n][k] bf16 (transposed)

    v8f acc[4];
#pragma unroll
    for (int i = 0; i < 4; ++i) acc[i] = zero8();

    for (int k0 = 0; k0 < 256; k0 += 32) {
        // A tile: 64 x 32 f32 -> bf16
        for (int e = tid; e < 512; e += 128) {
            int r = e >> 3, c = (e & 7) << 2;
            const float4 v = *(const float4*)&X[(size_t)(m0 + r) * 256 + k0 + c];
            Asm[r][c + 0] = f2bfu(v.x); Asm[r][c + 1] = f2bfu(v.y);
            Asm[r][c + 2] = f2bfu(v.z); Asm[r][c + 3] = f2bfu(v.w);
        }
        // B tile: W[k0..k0+32, nBase..nBase+64] -> transposed bf16
        for (int e = tid; e < 512; e += 128) {
            int r = e >> 4, c = (e & 15) << 2;
            const float4 v = *(const float4*)&Wp[(size_t)(k0 + r) * ldW + nBase + c];
            Bsm[c + 0][r] = f2bfu(v.x); Bsm[c + 1][r] = f2bfu(v.y);
            Bsm[c + 2][r] = f2bfu(v.z); Bsm[c + 3][r] = f2bfu(v.w);
        }
        __syncthreads();

        v16bf a = load_a_frag(&Asm[wid * 16 + laneLo][0], laneHi);
#pragma unroll
        for (int ni = 0; ni < 4; ++ni) {
            v16bf b = load_b_frag(&Bsm[ni * 16 + laneLo][0], laneHi);
            acc[ni] = wmma_bf16(a, b, acc[ni]);
        }
        __syncthreads();
    }

    // Epilogue: bias + scatter to shifted/windowed layout
#pragma unroll
    for (int ni = 0; ni < 4; ++ni) {
        int c = n0 + ni * 16 + laneLo;      // 0..767
        int head, dk; unsigned short* dst; float bias;
        if (c < 256)      { head = c >> 5;         dk = c & 31; dst = qw; bias = bq[c]; }
        else if (c < 512) { head = (c - 256) >> 5; dk = c & 31; dst = kw; bias = bkv[c - 256]; }
        else              { head = (c - 512) >> 5; dk = c & 31; dst = vw; bias = bkv[c - 256]; }
#pragma unroll
        for (int r = 0; r < 8; ++r) {
            int p = m0 + wid * 16 + r + 8 * laneHi;   // pixel index
            int b_ = p >> 14;
            int h  = (p >> 7) & 127;
            int w  = p & 127;
            int hs = (h + 124) & 127;                 // (h - SHIFT) mod 128
            int ws = (w + 124) & 127;
            int win = ((hs >> 3) << 4) + (ws >> 3);
            int t   = ((hs & 7) << 3) + (ws & 7);
            size_t di = ((((size_t)(b_ * 8 + head) * 256 + win) * 64 + t) * 32 + dk);
            dst[di] = f2bfu(acc[ni][r] + bias);
        }
    }
}

// ---------------------------------------------------------------------------
// Kernel 2: one wave per (b, head, window). 64x64 attention via bf16 WMMA.
// ---------------------------------------------------------------------------
__global__ __launch_bounds__(32) void swin_attn(
    const unsigned short* __restrict__ qw, const unsigned short* __restrict__ kw,
    const unsigned short* __restrict__ vw, const float* __restrict__ rel_table,
    unsigned short* __restrict__ attn_out)
{
    const int lane = threadIdx.x;
    const int blk  = blockIdx.x;          // (b*8+head)*256 + win
    const int win  = blk & 255;
    const int bh   = blk >> 8;
    const int head = bh & 7;
    const int b_   = bh >> 3;
    const int wy = win >> 4, wx = win & 15;
    const int laneHi = lane >> 4, laneLo = lane & 15;

    __shared__ __align__(16) unsigned short Qs[64][36];  // [t][dk]
    __shared__ __align__(16) unsigned short Ks[64][36];  // [t][dk]
    __shared__ __align__(16) unsigned short Vt[32][68];  // [dk][t] (transposed)
    __shared__ __align__(16) unsigned short Ps[64][68];  // [tq][tk]
    __shared__ float PEs[225];
    __shared__ int   labS[64];

    const size_t base = (size_t)blk * 64 * 32;
    const unsigned* qg = (const unsigned*)(qw + base);
    const unsigned* kg = (const unsigned*)(kw + base);
    const unsigned* vg = (const unsigned*)(vw + base);

    for (int e = lane; e < 1024; e += 32) {
        int row = e >> 4, c2 = e & 15;
        *(unsigned*)&Qs[row][c2 * 2] = qg[e];
        *(unsigned*)&Ks[row][c2 * 2] = kg[e];
        unsigned uv = vg[e];
        Vt[c2 * 2 + 0][row] = (unsigned short)(uv & 0xFFFFu);
        Vt[c2 * 2 + 1][row] = (unsigned short)(uv >> 16);
    }
    for (int e = lane; e < 225; e += 32) PEs[e] = rel_table[e * 8 + head];
    for (int e = lane; e < 64; e += 32) {
        int i = e >> 3, j = e & 7;
        int h = (wy * 8 + i + 4) & 127;
        int w = (wx * 8 + j + 4) & 127;
        int bhh = (h < 120) ? 0 : (h < 124 ? 1 : 2);
        int bww = (w < 120) ? 0 : (w < 124 ? 1 : 2);
        labS[e] = bhh * 3 + bww;
    }
    __syncthreads();

    // S = Q K^T : B-frag column n maps to K row (ni*16+n), contiguous dk run.
    v8f sc[4][4];
#pragma unroll
    for (int mi = 0; mi < 4; ++mi) {
        v16bf a = load_a_frag(&Qs[mi * 16 + laneLo][0], laneHi);
#pragma unroll
        for (int ni = 0; ni < 4; ++ni) {
            v16bf b = load_b_frag(&Ks[ni * 16 + laneLo][0], laneHi);
            sc[mi][ni] = wmma_bf16(a, b, zero8());
        }
    }

    // scale + rel-pos bias + mask, row softmax (rows span 16-lane half-groups)
    const float scale = 0.17677669529663687f;   // 1/sqrt(32)
#pragma unroll
    for (int mi = 0; mi < 4; ++mi) {
#pragma unroll
        for (int r = 0; r < 8; ++r) {
            int tq = mi * 16 + r + 8 * laneHi;
            int iq = tq >> 3, jq = tq & 7;
            int lq = labS[tq];
            float vals[4];
            float mx = -3.0e38f;
#pragma unroll
            for (int ni = 0; ni < 4; ++ni) {
                int tk = ni * 16 + laneLo;
                int ik = tk >> 3, jk = tk & 7;
                float s = sc[mi][ni][r] * scale + PEs[(iq - ik + 7) * 15 + (jq - jk + 7)];
                if (lq != labS[tk]) s = -1.0e9f;
                vals[ni] = s;
                mx = fmaxf(mx, s);
            }
#pragma unroll
            for (int off = 8; off >= 1; off >>= 1) mx = fmaxf(mx, __shfl_xor(mx, off));
            float sum = 0.0f;
#pragma unroll
            for (int ni = 0; ni < 4; ++ni) { vals[ni] = __expf(vals[ni] - mx); sum += vals[ni]; }
#pragma unroll
            for (int off = 8; off >= 1; off >>= 1) sum += __shfl_xor(sum, off);
            float inv = 1.0f / sum;
#pragma unroll
            for (int ni = 0; ni < 4; ++ni)
                Ps[tq][ni * 16 + laneLo] = f2bfu(vals[ni] * inv);
        }
    }
    __syncthreads();

    // O = P @ V (K-dim = 64 positions, two WMMA k-steps)
#pragma unroll
    for (int mi = 0; mi < 4; ++mi) {
        v16bf a0 = load_a_frag(&Ps[mi * 16 + laneLo][0],  laneHi);
        v16bf a1 = load_a_frag(&Ps[mi * 16 + laneLo][32], laneHi);
#pragma unroll
        for (int nd = 0; nd < 2; ++nd) {
            int dk = nd * 16 + laneLo;
            v16bf b0 = load_b_frag(&Vt[dk][0],  laneHi);
            v16bf b1 = load_b_frag(&Vt[dk][32], laneHi);
            v8f o = wmma_bf16(a0, b0, zero8());
            o = wmma_bf16(a1, b1, o);
#pragma unroll
            for (int r = 0; r < 8; ++r) {
                int t = mi * 16 + r + 8 * laneHi;
                int i = t >> 3, j = t & 7;
                int h = (wy * 8 + i + 4) & 127;       // un-shift on the way out
                int w = (wx * 8 + j + 4) & 127;
                size_t di = (((size_t)(b_ * 128 + h) * 128 + w) * 256) + head * 32 + dk;
                attn_out[di] = f2bfu(o[r]);
            }
        }
    }
}

// ---------------------------------------------------------------------------
// Kernel 3: output projection. attn(bf16)[131072,256] @ Wo + bo -> f32 out.
// ---------------------------------------------------------------------------
__global__ __launch_bounds__(128) void swin_out_gemm(
    const unsigned short* __restrict__ attn, const float* __restrict__ Wo,
    const float* __restrict__ bo, float* __restrict__ out)
{
    const int nt = blockIdx.x & 3;
    const int mt = blockIdx.x >> 2;
    const int n0 = nt * 64;
    const int m0 = mt * 64;
    const int tid = threadIdx.x;
    const int wid = tid >> 5, lane = tid & 31;
    const int laneHi = lane >> 4, laneLo = lane & 15;

    __shared__ __align__(16) unsigned short Asm[64][40];
    __shared__ __align__(16) unsigned short Bsm[64][40];

    v8f acc[4];
#pragma unroll
    for (int i = 0; i < 4; ++i) acc[i] = zero8();

    for (int k0 = 0; k0 < 256; k0 += 32) {
        for (int e = tid; e < 1024; e += 128) {
            int r = e >> 4, c2 = e & 15;
            *(unsigned*)&Asm[r][c2 * 2] =
                *(const unsigned*)&attn[(size_t)(m0 + r) * 256 + k0 + c2 * 2];
        }
        for (int e = tid; e < 512; e += 128) {
            int r = e >> 4, c = (e & 15) << 2;
            const float4 v = *(const float4*)&Wo[(size_t)(k0 + r) * 256 + n0 + c];
            Bsm[c + 0][r] = f2bfu(v.x); Bsm[c + 1][r] = f2bfu(v.y);
            Bsm[c + 2][r] = f2bfu(v.z); Bsm[c + 3][r] = f2bfu(v.w);
        }
        __syncthreads();

        v16bf a = load_a_frag(&Asm[wid * 16 + laneLo][0], laneHi);
#pragma unroll
        for (int ni = 0; ni < 4; ++ni) {
            v16bf b = load_b_frag(&Bsm[ni * 16 + laneLo][0], laneHi);
            acc[ni] = wmma_bf16(a, b, acc[ni]);
        }
        __syncthreads();
    }

#pragma unroll
    for (int ni = 0; ni < 4; ++ni) {
        int c = n0 + ni * 16 + laneLo;
        float bias = bo[c];
#pragma unroll
        for (int r = 0; r < 8; ++r) {
            int p = m0 + wid * 16 + r + 8 * laneHi;
            out[(size_t)p * 256 + c] = acc[ni][r] + bias;
        }
    }
}

extern "C" void kernel_launch(void* const* d_in, const int* in_sizes, int n_in,
                              void* d_out, int out_size, void* d_ws, size_t ws_size,
                              hipStream_t stream) {
    const float* x   = (const float*)d_in[0];
    const float* z   = (const float*)d_in[1];
    const float* Wq  = (const float*)d_in[2];
    const float* bq  = (const float*)d_in[3];
    const float* Wkv = (const float*)d_in[4];
    const float* bkv = (const float*)d_in[5];
    const float* Wo  = (const float*)d_in[6];
    const float* bo  = (const float*)d_in[7];
    const float* rel = (const float*)d_in[8];
    float* out = (float*)d_out;

    const size_t NE = (size_t)8 * 128 * 128 * 256;   // 33,554,432 elements
    unsigned short* qw   = (unsigned short*)d_ws;
    unsigned short* kw   = qw + NE;
    unsigned short* vw   = kw + NE;
    unsigned short* attn = vw + NE;

    swin_qkv_gemm<<<dim3(2048 * 12), dim3(128), 0, stream>>>(x, z, Wq, bq, Wkv, bkv, qw, kw, vw);
    swin_attn<<<dim3(16384), dim3(32), 0, stream>>>(qw, kw, vw, rel, attn);
    swin_out_gemm<<<dim3(2048 * 4), dim3(128), 0, stream>>>(attn, Wo, bo, out);
}
